// Encoder1_74491912782085
// MI455X (gfx1250) — matrix operand order repaired and checked by reading, not compiled, not run on hardware
//
#include <hip/hip_runtime.h>
#include <hip/hip_bf16.h>
#include <math.h>

// ---------------------------------------------------------------------------
// Types for CDNA5 WMMA (wave32): v16bf A/B operands, v8f f32 accumulator.
// ---------------------------------------------------------------------------
typedef __bf16 bf16_t;
typedef __bf16 v16bf __attribute__((ext_vector_type(16)));
typedef float  v8f   __attribute__((ext_vector_type(8)));

union BfPack { uint4 u[2]; v16bf v; };

// A tile (16x32, row-major source, ld = K stride in elements).
// Per ISA 7.12.2: lanes 0-15 hold rows M=0..15 with K=0..7 (V0..3) and
// K=16..23 (V4..7); lanes 16-31 hold K=8..15 and K=24..31.
__device__ __forceinline__ v16bf load_tile_a(const bf16_t* __restrict__ A,
                                             int ld, int m0, int k0, int lane) {
  int m   = m0 + (lane & 15);
  int klo = (lane & 16) ? 8 : 0;
  const bf16_t* p = A + (size_t)m * ld + k0 + klo;
  BfPack t;
  t.u[0] = *(const uint4*)(p);        // K = klo .. klo+7
  t.u[1] = *(const uint4*)(p + 16);   // K = klo+16 .. klo+23
  return t.v;
}

// B tile (32x16). B[k,n] = W[n,k] with W row-major (N x K): lane holds one
// output column n; lanes 0-15 hold K=0..7,16..23, lanes 16-31 hold K=8..15,24..31.
__device__ __forceinline__ v16bf load_tile_b(const bf16_t* __restrict__ W,
                                             int ld, int n0, int k0, int lane) {
  int n    = n0 + (lane & 15);
  int koff = (lane & 16) ? 8 : 0;
  const bf16_t* p = W + (size_t)n * ld + k0 + koff;
  BfPack t;
  t.u[0] = *(const uint4*)(p);        // K = koff .. koff+7
  t.u[1] = *(const uint4*)(p + 16);   // K = koff+16 .. koff+23
  return t.v;
}

__device__ __forceinline__ v8f wmma_bf16(v16bf a, v16bf b, v8f c) {
  // (neg_a, A, neg_b, B, c_mod, C, reuse_a, reuse_b)
  return __builtin_amdgcn_wmma_f32_16x16x32_bf16(false, a, false, b,
                                                 (short)0, c, false, false);
}

__device__ __forceinline__ float sigmoidf_fast(float x) {
  return 1.0f / (1.0f + __expf(-x));
}

// ---------------------------------------------------------------------------
// Embedding gather + f32 -> bf16 cast.  One block per (t,b) row of 512 elems.
// ---------------------------------------------------------------------------
__global__ void __launch_bounds__(128)
embed_cast_kernel(const int* __restrict__ src, const float* __restrict__ emb,
                  bf16_t* __restrict__ out) {
  int row = blockIdx.x;               // t*B + b
  int tok = src[row];
  const float4* e = (const float4*)(emb + (size_t)tok * 512);
  float4 v = e[threadIdx.x];
  union { bf16_t b[4]; uint2 u; } pk;
  pk.b[0] = (bf16_t)v.x; pk.b[1] = (bf16_t)v.y;
  pk.b[2] = (bf16_t)v.z; pk.b[3] = (bf16_t)v.w;
  ((uint2*)(out + (size_t)row * 512))[threadIdx.x] = pk.u;
}

// f32 -> bf16 weight conversion (vectorized by 4).
__global__ void f32_to_bf16_kernel(const float* __restrict__ in,
                                   bf16_t* __restrict__ out, int n4) {
  int i = blockIdx.x * blockDim.x + threadIdx.x;
  if (i >= n4) return;
  float4 v = ((const float4*)in)[i];
  union { bf16_t b[4]; uint2 u; } pk;
  pk.b[0] = (bf16_t)v.x; pk.b[1] = (bf16_t)v.y;
  pk.b[2] = (bf16_t)v.z; pk.b[3] = (bf16_t)v.w;
  ((uint2*)out)[i] = pk.u;
}

// Zero h state (bf16) and c state (f32).
__global__ void init_state_kernel(bf16_t* __restrict__ h, float* __restrict__ c,
                                  int n) {
  int i = blockIdx.x * blockDim.x + threadIdx.x;
  if (i < n) { h[i] = (bf16_t)0.0f; c[i] = 0.0f; }
}

// ---------------------------------------------------------------------------
// Big parallel GEMM: xW[m, g] = sum_k A[m,k] * W[g,k] + bih[g] + bhh[g]
//   A: (M=32768, 512) bf16   W: (2048, 512) bf16   out: (M, 2048) f32
// One wave per 16x16 output tile; 16 WMMAs (K=512).
// ---------------------------------------------------------------------------
__global__ void __launch_bounds__(256)
gemm_xw_kernel(const bf16_t* __restrict__ A, const bf16_t* __restrict__ W,
               const float* __restrict__ bih, const float* __restrict__ bhh,
               float* __restrict__ out) {
  int lane  = threadIdx.x & 31;
  int wave  = threadIdx.x >> 5;
  int tile  = blockIdx.x * 8 + wave;
  int ntile = tile & 127;             // 128 tiles across G=2048
  int mtile = tile >> 7;              // 2048 tiles across M=32768
  int m0 = mtile * 16, n0 = ntile * 16;

  v8f acc = {0.f, 0.f, 0.f, 0.f, 0.f, 0.f, 0.f, 0.f};
  for (int k0 = 0; k0 < 512; k0 += 32) {
    if (k0 + 32 < 512)  // hint next A chunk into cache (global_prefetch_b8)
      __builtin_prefetch(A + (size_t)(m0 + (lane & 15)) * 512 + k0 + 32, 0, 1);
    v16bf a = load_tile_a(A, 512, m0, k0, lane);
    v16bf b = load_tile_b(W, 512, n0, k0, lane);
    acc = wmma_bf16(a, b, acc);
  }

  int col   = n0 + (lane & 15);
  int rbase = m0 + ((lane & 16) ? 8 : 0);
  float bias = bih[col] + bhh[col];
#pragma unroll
  for (int i = 0; i < 8; ++i)
    out[(size_t)(rbase + i) * 2048 + col] = acc[i] + bias;
}

// ---------------------------------------------------------------------------
// One LSTM time step, fully fused:
//   gates = xW[t] + h_in @ Whh^T ; c' = sig(f)*c + sig(i)*tanh(g) ;
//   h' = sig(o)*tanh(c')
// 128 waves: wave = (mtile in 0..3) x (ntile in 0..31).  Each wave computes
// the 16x16 slice of all four gates (A tile shared by 4 WMMAs per k-step),
// then the elementwise update for its exclusive (row, col) region.
// ---------------------------------------------------------------------------
__global__ void __launch_bounds__(128)
lstm_step_kernel(const float* __restrict__ xw_t,    // 64 x 2048 (pre-offset by t)
                 const bf16_t* __restrict__ Whh,    // 2048 x 512 bf16
                 const bf16_t* __restrict__ h_in,   // 64 x 512 bf16
                 bf16_t* __restrict__ h_out,        // 64 x 512 bf16 (double buf)
                 float* __restrict__ c_st,          // 64 x 512 f32
                 bf16_t* __restrict__ hseq_t,       // layer0: 64x512 slice; else null
                 const int* __restrict__ sen_len,   // layer1: per-batch len; else null
                 float* __restrict__ final_out,     // layer1: 64x512 f32; else null
                 int t) {
  int lane  = threadIdx.x & 31;
  int wave  = blockIdx.x * (blockDim.x >> 5) + (threadIdx.x >> 5);  // 0..127
  int mtile = wave >> 5;              // 0..3   (batch rows)
  int ntile = wave & 31;              // 0..31  (hidden cols)
  int m0 = mtile * 16, n0 = ntile * 16;
  int col   = n0 + (lane & 15);
  int rbase = m0 + ((lane & 16) ? 8 : 0);

  // Init accumulators from precomputed xW (already includes both biases).
  v8f acc[4];
#pragma unroll
  for (int g = 0; g < 4; ++g)
#pragma unroll
    for (int i = 0; i < 8; ++i)
      acc[g][i] = xw_t[(size_t)(rbase + i) * 2048 + g * 512 + col];

  for (int k0 = 0; k0 < 512; k0 += 32) {
    v16bf a = load_tile_a(h_in, 512, m0, k0, lane);   // shared by all 4 gates
#pragma unroll
    for (int g = 0; g < 4; ++g) {
      v16bf b = load_tile_b(Whh, 512, g * 512 + n0, k0, lane);
      acc[g] = wmma_bf16(a, b, acc[g]);
    }
  }

#pragma unroll
  for (int i = 0; i < 8; ++i) {
    int row = rbase + i;                              // batch index
    float iv = sigmoidf_fast(acc[0][i]);
    float fv = sigmoidf_fast(acc[1][i]);
    float gv = tanhf(acc[2][i]);
    float ov = sigmoidf_fast(acc[3][i]);
    size_t idx = (size_t)row * 512 + col;
    float c = fv * c_st[idx] + iv * gv;
    c_st[idx] = c;
    float h = ov * tanhf(c);
    h_out[idx] = (bf16_t)h;
    if (hseq_t) hseq_t[idx] = (bf16_t)h;              // layer0 -> layer1 input
    if (final_out && (sen_len[row] - 1 == t)) final_out[idx] = h;
  }
}

// ---------------------------------------------------------------------------
// Host launch
// ---------------------------------------------------------------------------
extern "C" void kernel_launch(void* const* d_in, const int* in_sizes, int n_in,
                              void* d_out, int out_size, void* d_ws, size_t ws_size,
                              hipStream_t stream) {
  const int T = 512, B = 64, E = 512, H = 512, G = 4 * 512;

  const int*   src     = (const int*)d_in[0];
  const int*   sen_len = (const int*)d_in[1];
  const float* emb     = (const float*)d_in[2];
  const float* Wih0    = (const float*)d_in[3];
  const float* Whh0    = (const float*)d_in[4];
  const float* bih0    = (const float*)d_in[5];
  const float* bhh0    = (const float*)d_in[6];
  const float* Wih1    = (const float*)d_in[7];
  const float* Whh1    = (const float*)d_in[8];
  const float* bih1    = (const float*)d_in[9];
  const float* bhh1    = (const float*)d_in[10];

  // Workspace carve-up (~344 MB total).
  char* p = (char*)d_ws;
  auto carve = [&](size_t bytes) {
    char* r = p;
    p += (bytes + 255) & ~(size_t)255;
    return (void*)r;
  };
  bf16_t* ebf   = (bf16_t*)carve((size_t)T * B * E * 2);   // embedded, bf16
  bf16_t* h1seq = (bf16_t*)carve((size_t)T * B * H * 2);   // layer0 h sequence
  bf16_t* wih0b = (bf16_t*)carve((size_t)G * E * 2);
  bf16_t* whh0b = (bf16_t*)carve((size_t)G * H * 2);
  bf16_t* wih1b = (bf16_t*)carve((size_t)G * H * 2);
  bf16_t* whh1b = (bf16_t*)carve((size_t)G * H * 2);
  float*  xW    = (float*)carve((size_t)T * B * G * 4);    // 256 MB, reused per layer
  bf16_t* hA    = (bf16_t*)carve((size_t)B * H * 2);
  bf16_t* hB    = (bf16_t*)carve((size_t)B * H * 2);
  float*  cst   = (float*)carve((size_t)B * H * 4);

  const int wn4 = G * E / 4;
  const int gemm_blocks = (T * B / 16) * (G / 16) / 8;     // 32768

  // Embedding gather + casts.
  embed_cast_kernel<<<T * B, 128, 0, stream>>>(src, emb, ebf);
  f32_to_bf16_kernel<<<(wn4 + 255) / 256, 256, 0, stream>>>(Wih0, wih0b, wn4);
  f32_to_bf16_kernel<<<(wn4 + 255) / 256, 256, 0, stream>>>(Whh0, whh0b, wn4);
  f32_to_bf16_kernel<<<(wn4 + 255) / 256, 256, 0, stream>>>(Wih1, wih1b, wn4);
  f32_to_bf16_kernel<<<(wn4 + 255) / 256, 256, 0, stream>>>(Whh1, whh1b, wn4);

  // ----- Layer 0 -----
  gemm_xw_kernel<<<gemm_blocks, 256, 0, stream>>>(ebf, wih0b, bih0, bhh0, xW);
  init_state_kernel<<<(B * H + 255) / 256, 256, 0, stream>>>(hA, cst, B * H);
  {
    bf16_t* hin = hA; bf16_t* hout = hB;
    for (int t = 0; t < T; ++t) {
      lstm_step_kernel<<<32, 128, 0, stream>>>(
          xW + (size_t)t * B * G, whh0b, hin, hout, cst,
          h1seq + (size_t)t * B * H, nullptr, nullptr, t);
      bf16_t* tmp = hin; hin = hout; hout = tmp;
    }
  }

  // ----- Layer 1 -----
  gemm_xw_kernel<<<gemm_blocks, 256, 0, stream>>>(h1seq, wih1b, bih1, bhh1, xW);
  init_state_kernel<<<(B * H + 255) / 256, 256, 0, stream>>>(hA, cst, B * H);
  {
    bf16_t* hin = hA; bf16_t* hout = hB;
    for (int t = 0; t < T; ++t) {
      lstm_step_kernel<<<32, 128, 0, stream>>>(
          xW + (size_t)t * B * G, whh1b, hin, hout, cst,
          nullptr, sen_len, (float*)d_out, t);
      bf16_t* tmp = hin; hin = hout; hout = tmp;
    }
  }
}
